// DGMNet_84361747628698
// MI455X (gfx1250) — matrix operand order repaired and checked by reading, not compiled, not run on hardware
//
#include <hip/hip_runtime.h>

// ---------------------------------------------------------------------------
// DGM network forward, fused for MI455X (gfx1250, wave32, WMMA).
//  - All GEMMs on v_wmma_f32_16x16x32_f16 (f16 in, f32 accumulate).
//  - One workgroup = 128 rows; activation state lives in LDS for all 15
//    layers. LDS = x + S(ping) + S(pong) + S*R = 100 KB -> 3 WGs/WGP.
//  - Z/G gates kept in per-wave registers (packed f16 fragments); A-fragments
//    hoisted once per pass; x-fragment hoisted once per kernel.
//  - U/W0 weights zero-padded to K=32 so padded WMMAs are branch-free.
//  - tanh via 1 - 2*rcp(exp(2x)+1): ~5 VALU ops instead of libm tanhf.
// ---------------------------------------------------------------------------

typedef _Float16 half_t;
typedef __attribute__((ext_vector_type(16))) _Float16 v16h;
typedef __attribute__((ext_vector_type(8)))  _Float16 v8h;
typedef __attribute__((ext_vector_type(8)))  float    v8f;
typedef __attribute__((ext_vector_type(4)))  float    v4f;

#define WIDTH  128
#define TILE_M 128

union AFrag { v16h v; v8h h[2]; };

__device__ __forceinline__ v8f wmma_f16(v16h a, v16h b, v8f c)
{
    return __builtin_amdgcn_wmma_f32_16x16x32_f16(false, a, false, b,
                                                  (short)0, c, false, false);
}

// B fragment: transposed weights w[n][k], lane holds column n = nb+lane%16,
// K = kb + (lane/16)*8 .. +7 and kb + 16 + (lane/16)*8 .. +7 -> two 16B loads.
__device__ __forceinline__ v16h ldb(const half_t* w, int n, int kb,
                                    int kstride, int kh)
{
    const half_t* p = w + n * kstride + kb + (kh << 3);
    AFrag f;
    f.h[0] = *(const v8h*)(p);
    f.h[1] = *(const v8h*)(p + 16);
    return f.v;
}

// Fast tanh: saturates correctly for |x| large (exp->inf/0 -> +/-1).
__device__ __forceinline__ float ftanh(float x)
{
    float e = __expf(x + x);
    return 1.0f - 2.0f * __builtin_amdgcn_rcpf(e + 1.0f);
}

__global__ void __launch_bounds__(256)
dgm_fused(const float* __restrict__ x,      // [N][16]
          const float* __restrict__ b0,     // [128]
          const float* __restrict__ bgates, // [3][4][128]
          const float* __restrict__ bd,     // [2][128]
          const float* __restrict__ Wf,     // [128]
          const float* __restrict__ bf,     // [1]
          const half_t* __restrict__ w0t,   // [128][32]  (K zero-padded)
          const half_t* __restrict__ ut,    // [3][4][128][32]  (K zero-padded)
          const half_t* __restrict__ wt,    // [3][4][128][128]
          const half_t* __restrict__ wdt,   // [2][128][128]
          float* __restrict__ out)          // [N]
{
    extern __shared__ half_t smem[];
    half_t* sX = smem;                  // 128 x 16
    half_t* sBufA = smem + 2048;        // S ping
    half_t* sBufB = sBufA + 16384;      // S pong
    half_t* sP = sBufB + 16384;         // S*R product (H's K operand)

    const int  tid  = threadIdx.x;
    const int  wave = tid >> 5;
    const int  lane = tid & 31;
    const int  ln   = lane & 15;
    const int  kh   = lane >> 4;        // lane's K-half
    const int  mr   = wave << 4;        // wave's 16-row strip
    const int  mrow = mr + ln;          // A-fragment row
    const int  mb   = mr + (kh << 3);   // C/D-fragment row base
    const long tile = (long)blockIdx.x * TILE_M;

    // ---- stage x tile to LDS as f16 (2048 floats, 8 per thread) ----
    {
        const float* xp = x + tile * 16 + tid * 8;
        v4f x0 = *(const v4f*)(xp);
        v4f x1 = *(const v4f*)(xp + 4);
        half_t* d = sX + tid * 8;
#pragma unroll
        for (int q = 0; q < 4; ++q) { d[q] = (half_t)x0[q]; d[4 + q] = (half_t)x1[q]; }
    }
    __syncthreads();

    // x A-fragment (K=16 valid, K=16..31 zero): held in regs for whole kernel
    AFrag ax{};
    ax.h[0] = *(const v8h*)(sX + mrow * 16 + (kh << 3));

    half_t* sS = sBufA;
    half_t* sN = sBufB;

    // ---- initial dense layer: S = relu(x@W0 + b0) ----
#pragma unroll
    for (int c = 0; c < 8; ++c) {
        const int n = (c << 4) + ln;
        v8f acc = {};
        acc = wmma_f16(ax.v, ldb(w0t, n, 0, 32, kh), acc);
        const float bv = b0[n];
#pragma unroll
        for (int r = 0; r < 8; ++r)
            sS[(mb + r) * WIDTH + n] = (half_t)fmaxf(acc[r] + bv, 0.0f);
    }
    __syncthreads();

    // ---- 3 DGM LSTM-style layers ----
    for (int i = 0; i < 3; ++i) {
        const half_t* Uz = ut + (i * 4 + 0) * 4096;
        const half_t* Ug = ut + (i * 4 + 1) * 4096;
        const half_t* Ur = ut + (i * 4 + 2) * 4096;
        const half_t* Uh = ut + (i * 4 + 3) * 4096;
        const half_t* Wz = wt + (i * 4 + 0) * 16384;
        const half_t* Wg = wt + (i * 4 + 1) * 16384;
        const half_t* Wr = wt + (i * 4 + 2) * 16384;
        const half_t* Wh = wt + (i * 4 + 3) * 16384;
        const float*  bi = bgates + i * 512;

        // hoist S A-fragments once for the whole Z/G/R pass
        AFrag as[4];
#pragma unroll
        for (int kk = 0; kk < 4; ++kk) {
            const half_t* p = sS + mrow * WIDTH + (kk << 5) + (kh << 3);
            as[kk].h[0] = *(const v8h*)(p);
            as[kk].h[1] = *(const v8h*)(p + 16);
        }

        v8h zreg[8], greg[8];   // Z and G stay in registers (fragment layout)

        // ---- pass 1: Z, G, R in one sweep (A-fragments reused 3x) ----
#pragma unroll
        for (int c = 0; c < 8; ++c) {
            const int n = (c << 4) + ln;
            v8f az = {}, ag = {}, ar = {};
            az = wmma_f16(ax.v, ldb(Uz, n, 0, 32, kh), az);
            ag = wmma_f16(ax.v, ldb(Ug, n, 0, 32, kh), ag);
            ar = wmma_f16(ax.v, ldb(Ur, n, 0, 32, kh), ar);
#pragma unroll
            for (int kk = 0; kk < 4; ++kk) {
                az = wmma_f16(as[kk].v, ldb(Wz, n, kk << 5, WIDTH, kh), az);
                ag = wmma_f16(as[kk].v, ldb(Wg, n, kk << 5, WIDTH, kh), ag);
                ar = wmma_f16(as[kk].v, ldb(Wr, n, kk << 5, WIDTH, kh), ar);
            }
            const float bz = bi[n], bgv = bi[128 + n], br = bi[256 + n];
#pragma unroll
            for (int r = 0; r < 8; ++r) {
                zreg[c][r] = (half_t)ftanh(az[r] + bz);
                greg[c][r] = (half_t)ftanh(ag[r] + bgv);
                const int idx = (mb + r) * WIDTH + n;
                sP[idx] = (half_t)(ftanh(ar[r] + br) * (float)sS[idx]);
            }
        }
        __syncthreads();   // all rows of sP needed as K operand below

        // ---- pass 2: H = tanh(x@Uh + (S*R)@Wh), S' = (1-G)*H + Z*S ----
        AFrag ap[4];
#pragma unroll
        for (int kk = 0; kk < 4; ++kk) {
            const half_t* p = sP + mrow * WIDTH + (kk << 5) + (kh << 3);
            ap[kk].h[0] = *(const v8h*)(p);
            ap[kk].h[1] = *(const v8h*)(p + 16);
        }
#pragma unroll
        for (int c = 0; c < 8; ++c) {
            const int n = (c << 4) + ln;
            v8f ah = {};
            ah = wmma_f16(ax.v, ldb(Uh, n, 0, 32, kh), ah);
#pragma unroll
            for (int kk = 0; kk < 4; ++kk)
                ah = wmma_f16(ap[kk].v, ldb(Wh, n, kk << 5, WIDTH, kh), ah);
            const float bh = bi[384 + n];
#pragma unroll
            for (int r = 0; r < 8; ++r) {
                const int idx = (mb + r) * WIDTH + n;
                float h = ftanh(ah[r] + bh);
                float z = (float)zreg[c][r];
                float g = (float)greg[c][r];
                float s = (float)sS[idx];
                sN[idx] = (half_t)((1.0f - g) * h + z * s);
            }
        }
        __syncthreads();
        half_t* t = sS; sS = sN; sN = t;
    }

    // ---- 2 dense tanh layers ----
    for (int j = 0; j < 2; ++j) {
        const half_t* Wj = wdt + j * 16384;
        const float*  bj = bd + j * 128;
        AFrag af[4];
#pragma unroll
        for (int kk = 0; kk < 4; ++kk) {
            const half_t* p = sS + mrow * WIDTH + (kk << 5) + (kh << 3);
            af[kk].h[0] = *(const v8h*)(p);
            af[kk].h[1] = *(const v8h*)(p + 16);
        }
#pragma unroll
        for (int c = 0; c < 8; ++c) {
            const int n = (c << 4) + ln;
            v8f acc = {};
#pragma unroll
            for (int kk = 0; kk < 4; ++kk)
                acc = wmma_f16(af[kk].v, ldb(Wj, n, kk << 5, WIDTH, kh), acc);
            const float bv = bj[n];
#pragma unroll
            for (int r = 0; r < 8; ++r)
                sN[(mb + r) * WIDTH + n] = (half_t)ftanh(acc[r] + bv);
        }
        __syncthreads();
        half_t* t = sS; sS = sN; sN = t;
    }

    // ---- final linear layer: out = S @ Wf + bf (one f32 dot per row) ----
    if (tid < TILE_M) {
        float acc = bf[0];
        const half_t* row = sS + tid * WIDTH;
#pragma unroll 8
        for (int k = 0; k < WIDTH; ++k) acc += (float)row[k] * Wf[k];
        out[tile + tid] = acc;
    }
}

// ---------------------------------------------------------------------------
// One-shot weight conversion: f32 row-major [k][n] -> f16 transposed [n][k].
// W0/U are zero-padded from K=16 to K=32 so the padded WMMA path needs no
// divergent loads. Total 282624 halves (~565 KB) in d_ws.
// ---------------------------------------------------------------------------
__global__ void __launch_bounds__(256)
prep_weights(const float* __restrict__ W0, const float* __restrict__ U,
             const float* __restrict__ W,  const float* __restrict__ Wd,
             half_t* __restrict__ w0t, half_t* __restrict__ ut,
             half_t* __restrict__ wt,  half_t* __restrict__ wdt)
{
    const int i = blockIdx.x * 256 + threadIdx.x;
    if (i < 4096) {                                   // W0: 16x128 -> [128][32]
        int n = i >> 5, k = i & 31;
        w0t[i] = (k < 16) ? (half_t)W0[k * 128 + n] : (half_t)0.0f;
    }
    if (i < 49152) {                                  // U: 12 x 16x128 -> [128][32]
        int m = i >> 12, r = i & 4095, n = r >> 5, k = r & 31;
        ut[i] = (k < 16) ? (half_t)U[m * 2048 + k * 128 + n] : (half_t)0.0f;
    }
    if (i < 196608) {                                 // W: 12 x 128x128
        int m = i >> 14, r = i & 16383, n = r >> 7, k = r & 127;
        wt[i] = (half_t)W[m * 16384 + k * 128 + n];
    }
    if (i < 32768) {                                  // Wd: 2 x 128x128
        int m = i >> 14, r = i & 16383, n = r >> 7, k = r & 127;
        wdt[i] = (half_t)Wd[m * 16384 + k * 128 + n];
    }
}

extern "C" void kernel_launch(void* const* d_in, const int* in_sizes, int n_in,
                              void* d_out, int out_size, void* d_ws, size_t ws_size,
                              hipStream_t stream)
{
    const float* x  = (const float*)d_in[0];
    const float* W0 = (const float*)d_in[1];
    const float* b0 = (const float*)d_in[2];
    const float* U  = (const float*)d_in[3];
    const float* W  = (const float*)d_in[4];
    const float* b  = (const float*)d_in[5];
    const float* Wd = (const float*)d_in[6];
    const float* bd = (const float*)d_in[7];
    const float* Wf = (const float*)d_in[8];
    const float* bf = (const float*)d_in[9];
    float* out = (float*)d_out;

    half_t* ws  = (half_t*)d_ws;
    half_t* w0t = ws;                 //   4096 halves (padded K=32)
    half_t* ut  = w0t + 4096;         //  49152 halves (padded K=32)
    half_t* wt  = ut + 49152;         // 196608 halves
    half_t* wdt = wt + 196608;        //  32768 halves (total ~565 KB)

    prep_weights<<<768, 256, 0, stream>>>(W0, U, W, Wd, w0t, ut, wt, wdt);

    const int N      = in_sizes[0] / 16;       // 524288
    const int nTiles = N / TILE_M;             // 4096 workgroups
    const size_t shmem = (size_t)(2048 + 3 * 16384) * sizeof(half_t); // 100 KB

    dgm_fused<<<nTiles, 256, shmem, stream>>>(x, b0, b, bd, Wf, bf,
                                              w0t, ut, wt, wdt, out);
}
